// RSSM_60404420051349
// MI455X (gfx1250) — matrix operand order repaired and checked by reading, not compiled, not run on hardware
//
#include <hip/hip_runtime.h>
#include <hip/hip_bf16.h>

// ---------------------------------------------------------------------------
// RSSM obs_step scan for MI455X (gfx1250, wave32, WMMA bf16 16x16x32).
// B=1024, T=64, E=1024, A=6, S=30, D=H=200.
// 64 blocks, each owns 16 batch rows for all 64 timesteps (rows independent).
// Weights packed once to bf16 WMMA B-fragments in ws (L2-resident, ~1.2MB).
// embed pre-converted to bf16 in ws and staged per-step into LDS with
// GLOBAL_LOAD_ASYNC_TO_LDS_B128 (overlaps the emb+GRU GEMMs), when ws allows.
// ---------------------------------------------------------------------------

typedef __attribute__((ext_vector_type(16))) __bf16 v16bf;
typedef __attribute__((ext_vector_type(8)))  float  v8f;

union ABFrag { uint4 q[2]; v16bf v; };

#define NB   1024
#define NT_T 64
#define NE   1024
#define NA   6
#define NS   30
#define ND   200
#define NH   200

// Padded GEMM geometry (K -> mult of 32, N -> mult of 16)
#define KCH_EMB 2
#define TIL_EMB 13   // N 200 -> 208
#define KCH_GRU 13   // K 400 -> 416
#define TIL_GRU 38   // N 600 -> 608
#define KCH_PR1 7    // K 200 -> 224
#define TIL_PR1 13
#define KCH_PR2 7
#define TIL_PR2 4    // N 60 -> 64
#define KCH_PO1 39   // K 1224 -> 1248
#define TIL_PO1 13
#define KCH_PO2 7
#define TIL_PO2 4

// Workspace offsets (in halfs) for packed bf16 weight fragments
constexpr size_t OFF_W_EMB = 0;
constexpr size_t OFF_W_GRU = OFF_W_EMB + (size_t)KCH_EMB * TIL_EMB * 512;
constexpr size_t OFF_W_PR1 = OFF_W_GRU + (size_t)KCH_GRU * TIL_GRU * 512;
constexpr size_t OFF_W_PR2 = OFF_W_PR1 + (size_t)KCH_PR1 * TIL_PR1 * 512;
constexpr size_t OFF_W_PO1 = OFF_W_PR2 + (size_t)KCH_PR2 * TIL_PR2 * 512;
constexpr size_t OFF_W_PO2 = OFF_W_PO1 + (size_t)KCH_PO1 * TIL_PO1 * 512;
constexpr size_t W_HALFS_TOTAL = OFF_W_PO2 + (size_t)KCH_PO2 * TIL_PO2 * 512;

// Padded bias offsets (floats, relative to bias base in ws)
#define BI_EMB 0
#define BI_GRU 208
#define BI_PR1 816
#define BI_PR2 1024
#define BI_PO1 1088
#define BI_PO2 1296
#define BI_TOTAL 1360

// ws byte layout: [weights bf16][bias f32][embed bf16 (optional)]
constexpr size_t OFF_BIAS_BYTES  = W_HALFS_TOTAL * 2;                 // 1,202,176
constexpr size_t OFF_EMBBF_BYTES = OFF_BIAS_BYTES + BI_TOTAL * 4;     // 1,207,616 (16B aligned)
constexpr size_t EMBBF_BYTES     = (size_t)NB * NT_T * NE * 2;        // 128 MiB
constexpr size_t WS_NEED_ASYNC   = OFF_EMBBF_BYTES + EMBBF_BYTES;

// LDS layout (bytes)
#define L_PARTS   0                    // f32 [16][608]
#define L_DETER   38912                // f32 [16][200]
#define L_STOCH   51712                // f32 [16][32]
#define L_SPR     53760                // f32 [16][64]
#define L_SPO     57856                // f32 [16][64]
#define L_AEMB    61952                // u16 [16][64]
#define L_AGRU    64000                // u16 [16][416]
#define L_APR     77312                // u16 [16][224]
#define L_AH      84480                // u16 [16][224]
#define L_AH2     91648                // u16 [16][224]
#define L_APO     98816                // u16 [16][1248]
#define SMEM_BYTES 138752

__device__ inline unsigned short f2bf(float f) {
  unsigned u = __float_as_uint(f);
  u += 0x7FFFu + ((u >> 16) & 1u);            // round-to-nearest-even
  return (unsigned short)(u >> 16);
}

__device__ inline float sigmf(float x) { return 1.f / (1.f + __expf(-x)); }

__device__ inline unsigned pcg(unsigned x) {
  x = x * 747796405u + 2891336453u;
  unsigned w = ((x >> ((x >> 28) + 4u)) ^ x) * 277803737u;
  return (w >> 22) ^ w;
}

// deterministic standard normal, indexed by element + stream
__device__ inline float gauss(unsigned idx, unsigned salt) {
  unsigned a = pcg(idx * 2u + 1u + salt * 0x9E3779B9u);
  unsigned b = pcg(a ^ 0x85EBCA6Bu);
  float u1 = (float)(a >> 8) * (1.0f / 16777216.f) + 1e-7f;
  float u2 = (float)(b >> 8) * (1.0f / 16777216.f);
  return sqrtf(-2.f * __logf(u1)) * __cosf(6.28318530718f * u2);
}

// ---- GEMM over 16x16 output tiles, K in chunks of 32, bf16 WMMA ------------
// A: LDS row-major bf16 [16][KP].  W: packed fragments (tile-major, 512 halfs
// per (kc,nt) fragment, lane-major: frag[lane*16+h] = W[kc*32+((h&8)<<1)+(lane>>4)*8+(h&7)]
// [nt*16+(lane&15)]).  Epilogue fuses bias (+ELU) and writes bf16 (into the
// next stage's A buffer) or f32.

template <int KP, int KCH, int NTIL, int DSTRIDE>
__device__ inline void gemm_bias_elu_bf16(const unsigned short* __restrict__ aBuf,
                                          const unsigned short* __restrict__ wfr,
                                          const float* __restrict__ bias, int nValid,
                                          unsigned short* __restrict__ dst,
                                          int w0, int nw, int lane) {
  const int m0 = (lane >> 4) << 3;
  const int nl = lane & 15;
  const unsigned short* arow = aBuf + nl * KP + ((lane >> 4) << 3);
  for (int tt = w0; tt < NTIL; tt += nw) {
    if (tt + nw < NTIL) __builtin_prefetch(wfr + (size_t)(tt + nw) * KCH * 512, 0, 3);
    v8f acc = {};
    const unsigned short* wt = wfr + (size_t)tt * KCH * 512 + lane * 16;
#pragma unroll
    for (int kc = 0; kc < KCH; ++kc) {
      ABFrag a, b;
      const unsigned short* ap = arow + kc * 32;
      a.q[0] = *reinterpret_cast<const uint4*>(ap);
      a.q[1] = *reinterpret_cast<const uint4*>(ap + 16);
      const unsigned short* bp = wt + kc * 512;
      b.q[0] = *reinterpret_cast<const uint4*>(bp);
      b.q[1] = *reinterpret_cast<const uint4*>(bp + 8);
      acc = __builtin_amdgcn_wmma_f32_16x16x32_bf16(false, a.v, false, b.v,
                                                    (short)0, acc, false, false);
    }
    const int n = tt * 16 + nl;
    const float bv = bias[n];
#pragma unroll
    for (int r = 0; r < 8; ++r) {
      float v = acc[r] + bv;
      v = v > 0.f ? v : (__expf(v) - 1.f);   // ELU
      if (n < nValid) dst[(m0 + r) * DSTRIDE + n] = f2bf(v);
    }
  }
}

template <int KP, int KCH, int NTIL, int DSTRIDE>
__device__ inline void gemm_bias_f32(const unsigned short* __restrict__ aBuf,
                                     const unsigned short* __restrict__ wfr,
                                     const float* __restrict__ bias,
                                     float* __restrict__ dst,
                                     int w0, int nw, int lane) {
  const int m0 = (lane >> 4) << 3;
  const int nl = lane & 15;
  const unsigned short* arow = aBuf + nl * KP + ((lane >> 4) << 3);
  for (int tt = w0; tt < NTIL; tt += nw) {
    if (tt + nw < NTIL) __builtin_prefetch(wfr + (size_t)(tt + nw) * KCH * 512, 0, 3);
    v8f acc = {};
    const unsigned short* wt = wfr + (size_t)tt * KCH * 512 + lane * 16;
#pragma unroll
    for (int kc = 0; kc < KCH; ++kc) {
      ABFrag a, b;
      const unsigned short* ap = arow + kc * 32;
      a.q[0] = *reinterpret_cast<const uint4*>(ap);
      a.q[1] = *reinterpret_cast<const uint4*>(ap + 16);
      const unsigned short* bp = wt + kc * 512;
      b.q[0] = *reinterpret_cast<const uint4*>(bp);
      b.q[1] = *reinterpret_cast<const uint4*>(bp + 8);
      acc = __builtin_amdgcn_wmma_f32_16x16x32_bf16(false, a.v, false, b.v,
                                                    (short)0, acc, false, false);
    }
    const int n = tt * 16 + nl;
    const float bv = bias[n];
#pragma unroll
    for (int r = 0; r < 8; ++r) dst[(m0 + r) * DSTRIDE + n] = acc[r] + bv;
  }
}

// ---- weight pack: fp32 row-major [K][N] -> bf16 WMMA B-fragments -----------
__global__ void pack_weight(const float* __restrict__ W, int K, int N,
                            int KCH, int NTIL, unsigned short* __restrict__ dst) {
  const int total = KCH * NTIL * 512;
  for (int i = blockIdx.x * blockDim.x + threadIdx.x; i < total;
       i += gridDim.x * blockDim.x) {
    const int f = i >> 9, e = i & 511;
    const int lane = e >> 4, h = e & 15;
    const int kc = f % KCH, nt = f / KCH;
    const int k = kc * 32 + ((h & 8) << 1) + ((lane >> 4) << 3) + (h & 7);
    const int n = nt * 16 + (lane & 15);
    const float v = (k < K && n < N) ? W[(size_t)k * N + n] : 0.f;
    dst[i] = f2bf(v);
  }
}

__global__ void pack_bias(const float* __restrict__ src, int n,
                          float* __restrict__ dst, int npad) {
  const int i = blockIdx.x * blockDim.x + threadIdx.x;
  if (i < npad) dst[i] = (i < n) ? src[i] : 0.f;
}

// ---- one-shot streaming fp32 -> bf16 of the embed tensor (L2-resident) -----
__global__ void embed_bf16(const float* __restrict__ src,
                           unsigned short* __restrict__ dst, int n4) {
  const int i = blockIdx.x * blockDim.x + threadIdx.x;
  if (i < n4) {
    const float4 v = reinterpret_cast<const float4*>(src)[i];
    ushort4 o;
    o.x = f2bf(v.x); o.y = f2bf(v.y); o.z = f2bf(v.z); o.w = f2bf(v.w);
    reinterpret_cast<ushort4*>(dst)[i] = o;
  }
}

// ---- main persistent-per-block scan kernel ---------------------------------
template <bool ASYNC>
__global__ __launch_bounds__(256) void rssm_scan(
    const float* __restrict__ embed, const unsigned short* __restrict__ embBf,
    const float* __restrict__ action,
    const unsigned short* __restrict__ wf, const float* __restrict__ wb,
    float* __restrict__ out) {
  extern __shared__ char smem[];
  float*          parts = (float*)(smem + L_PARTS);
  float*          deter = (float*)(smem + L_DETER);
  float*          stoch = (float*)(smem + L_STOCH);
  float*          s_pr  = (float*)(smem + L_SPR);
  float*          s_po  = (float*)(smem + L_SPO);
  unsigned short* aEmb  = (unsigned short*)(smem + L_AEMB);
  unsigned short* aGru  = (unsigned short*)(smem + L_AGRU);
  unsigned short* aPr   = (unsigned short*)(smem + L_APR);
  unsigned short* aH    = (unsigned short*)(smem + L_AH);
  unsigned short* aH2   = (unsigned short*)(smem + L_AH2);
  unsigned short* aPo   = (unsigned short*)(smem + L_APO);

  const int tid = threadIdx.x, lane = tid & 31, wave = tid >> 5;
  const int b0 = blockIdx.x * 16;

  // zero carry + all K/N padding regions once
  for (int i = tid; i < SMEM_BYTES / 4; i += 256) ((unsigned*)smem)[i] = 0u;
  __syncthreads();

  for (int t = 0; t < NT_T; ++t) {
    // ---- Phase A: stage step inputs ----------------------------------------
    if constexpr (ASYNC) {
      // embed_t (bf16) -> aPo[:,200:1224] via async LDS DMA; 2048 x 16B.
      // Completion is only needed before the po1 GEMM (waited in Phase D),
      // so this overlaps the emb and GRU GEMMs.
#pragma unroll
      for (int i = 0; i < 8; ++i) {
        const int c = tid + i * 256;
        const int m = c >> 7;                 // 128 16B-chunks per row
        const int ko = (c & 127) << 3;        // half offset within row
        const unsigned long long gaddr = (unsigned long long)(uintptr_t)(
            embBf + ((size_t)(b0 + m) * NT_T + t) * NE + ko);
        const unsigned laddr =
            (unsigned)(uintptr_t)(aPo + m * 1248 + ND + ko);
        asm volatile("global_load_async_to_lds_b128 %0, %1, off"
                     :: "v"(laddr), "v"(gaddr) : "memory");
      }
    } else {
      for (int i = tid; i < 16 * NE; i += 256) {
        const int m = i >> 10, k = i & 1023;
        aPo[m * 1248 + ND + k] =
            f2bf(embed[((size_t)(b0 + m) * NT_T + t) * NE + k]);
      }
    }
    for (int i = tid; i < 16 * 64; i += 256) {            // [stoch|action] bf16
      const int m = i >> 6, k = i & 63;
      float v = 0.f;
      if (k < NS)           v = stoch[m * 32 + k];
      else if (k < NS + NA) v = action[((size_t)(b0 + m) * NT_T + t) * NA + (k - NS)];
      aEmb[i] = f2bf(v);
    }
    for (int i = tid; i < 16 * ND; i += 256) {            // prev deter -> aGru[:,200:]
      const int m = i / ND, j = i - m * ND;
      aGru[m * 416 + ND + j] = f2bf(deter[m * ND + j]);
    }
    __syncthreads();

    // ---- x = elu([stoch,a] @ w_emb + b) -> aGru[:,0:200] (bf16) ------------
    gemm_bias_elu_bf16<64, KCH_EMB, TIL_EMB, 416>(aEmb, wf + OFF_W_EMB,
        wb + BI_EMB, ND, aGru, wave, 8, lane);
    __syncthreads();

    // ---- parts = [x,deter] @ w_gru + b  (f32) ------------------------------
    gemm_bias_f32<416, KCH_GRU, TIL_GRU, 608>(aGru, wf + OFF_W_GRU,
        wb + BI_GRU, parts, wave, 8, lane);
    __syncthreads();

    // ---- GRU cell (elementwise) -> deter; bf16 copies to aPr / aPo[:,0:200]
    for (int i = tid; i < 16 * ND; i += 256) {
      const int m = i / ND, j = i - m * ND;
      const float* pr = parts + m * 608;
      const float r = sigmf(pr[j]);
      const float c = tanhf(r * pr[ND + j]);
      const float u = sigmf(pr[2 * ND + j] - 1.f);
      const float d = u * c + (1.f - u) * deter[m * ND + j];
      deter[m * ND + j] = d;
      const unsigned short hb = f2bf(d);
      aPr[m * 224 + j] = hb;
      aPo[m * 1248 + j] = hb;
    }
    if constexpr (ASYNC)                     // embed tile must be in LDS now
      asm volatile("s_wait_asynccnt 0x0" ::: "memory");
    __syncthreads();

    // ---- prior/posterior hidden layers (waves split 1:7 by chunk-tiles) ----
    if (wave < 1)
      gemm_bias_elu_bf16<224, KCH_PR1, TIL_PR1, 224>(aPr, wf + OFF_W_PR1,
          wb + BI_PR1, NH, aH, wave, 1, lane);
    else
      gemm_bias_elu_bf16<1248, KCH_PO1, TIL_PO1, 224>(aPo, wf + OFF_W_PO1,
          wb + BI_PO1, NH, aH2, wave - 1, 7, lane);
    __syncthreads();

    // ---- sufficient-stat heads (4 waves each, concurrent) ------------------
    if (wave < 4)
      gemm_bias_f32<224, KCH_PR2, TIL_PR2, 64>(aH, wf + OFF_W_PR2,
          wb + BI_PR2, s_pr, wave, 4, lane);
    else
      gemm_bias_f32<224, KCH_PO2, TIL_PO2, 64>(aH2, wf + OFF_W_PO2,
          wb + BI_PO2, s_po, wave - 4, 4, lane);
    __syncthreads();

    // ---- sample, pack outputs, update stoch carry --------------------------
    for (int i = tid; i < 16 * NS; i += 256) {
      const int m = i / NS, s = i - m * NS;
      const size_t base = ((size_t)(b0 + m) * NT_T + t) * 580;
      const float pm = s_pr[m * 64 + s];
      const float ps = 2.f * sigmf(0.5f * s_pr[m * 64 + NS + s]) + 0.1f;
      const float qm = s_po[m * 64 + s];
      const float qs = 2.f * sigmf(0.5f * s_po[m * 64 + NS + s]) + 0.1f;
      const unsigned gidx = ((unsigned)t * (unsigned)NB + (unsigned)(b0 + m)) * NS + s;
      const float pst = pm + ps * gauss(gidx, 0u);
      const float qst = qm + qs * gauss(gidx, 1u);
      out[base + s]       = qm;
      out[base + 30 + s]  = qs;
      out[base + 60 + s]  = qst;
      out[base + 290 + s] = pm;
      out[base + 320 + s] = ps;
      out[base + 350 + s] = pst;
      stoch[m * 32 + s] = qst;               // posterior sample is next carry
    }
    for (int i = tid; i < 16 * ND; i += 256) {
      const int m = i / ND, j = i - m * ND;
      const size_t base = ((size_t)(b0 + m) * NT_T + t) * 580;
      const float d = deter[m * ND + j];
      out[base + 90 + j]  = d;
      out[base + 380 + j] = d;
    }
    __syncthreads();
  }
}

extern "C" void kernel_launch(void* const* d_in, const int* in_sizes, int n_in,
                              void* d_out, int out_size, void* d_ws, size_t ws_size,
                              hipStream_t stream) {
  const float* embed  = (const float*)d_in[0];
  const float* action = (const float*)d_in[1];
  const float* w_emb  = (const float*)d_in[3];
  const float* b_emb  = (const float*)d_in[4];
  const float* w_gru  = (const float*)d_in[5];
  const float* b_gru  = (const float*)d_in[6];
  const float* w_pr1  = (const float*)d_in[7];
  const float* b_pr1  = (const float*)d_in[8];
  const float* w_pr2  = (const float*)d_in[9];
  const float* b_pr2  = (const float*)d_in[10];
  const float* w_po1  = (const float*)d_in[11];
  const float* b_po1  = (const float*)d_in[12];
  const float* w_po2  = (const float*)d_in[13];
  const float* b_po2  = (const float*)d_in[14];
  float* out = (float*)d_out;

  unsigned short* wsH   = (unsigned short*)d_ws;
  float*          wsB   = (float*)((char*)d_ws + OFF_BIAS_BYTES);
  unsigned short* embBf = (unsigned short*)((char*)d_ws + OFF_EMBBF_BYTES);

  auto gblk = [](int elems) { return (elems + 255) / 256; };

  // weight/bias packing (re-run every call: deterministic)
  pack_weight<<<gblk(KCH_EMB * TIL_EMB * 512), 256, 0, stream>>>(w_emb, 36, 200, KCH_EMB, TIL_EMB, wsH + OFF_W_EMB);
  pack_weight<<<gblk(KCH_GRU * TIL_GRU * 512), 256, 0, stream>>>(w_gru, 400, 600, KCH_GRU, TIL_GRU, wsH + OFF_W_GRU);
  pack_weight<<<gblk(KCH_PR1 * TIL_PR1 * 512), 256, 0, stream>>>(w_pr1, 200, 200, KCH_PR1, TIL_PR1, wsH + OFF_W_PR1);
  pack_weight<<<gblk(KCH_PR2 * TIL_PR2 * 512), 256, 0, stream>>>(w_pr2, 200, 60, KCH_PR2, TIL_PR2, wsH + OFF_W_PR2);
  pack_weight<<<gblk(KCH_PO1 * TIL_PO1 * 512), 256, 0, stream>>>(w_po1, 1224, 200, KCH_PO1, TIL_PO1, wsH + OFF_W_PO1);
  pack_weight<<<gblk(KCH_PO2 * TIL_PO2 * 512), 256, 0, stream>>>(w_po2, 200, 60, KCH_PO2, TIL_PO2, wsH + OFF_W_PO2);
  pack_bias<<<1, 256, 0, stream>>>(b_emb, 200, wsB + BI_EMB, 208);
  pack_bias<<<3, 256, 0, stream>>>(b_gru, 600, wsB + BI_GRU, 608);
  pack_bias<<<1, 256, 0, stream>>>(b_pr1, 200, wsB + BI_PR1, 208);
  pack_bias<<<1, 256, 0, stream>>>(b_pr2, 60,  wsB + BI_PR2, 64);
  pack_bias<<<1, 256, 0, stream>>>(b_po1, 200, wsB + BI_PO1, 208);
  pack_bias<<<1, 256, 0, stream>>>(b_po2, 60,  wsB + BI_PO2, 64);

  const bool asyncPath = (ws_size >= WS_NEED_ASYNC);
  if (asyncPath) {
    const int n4 = (int)((size_t)NB * NT_T * NE / 4);
    embed_bf16<<<gblk(n4), 256, 0, stream>>>(embed, embBf, n4);
    (void)hipFuncSetAttribute((const void*)rssm_scan<true>,
                              hipFuncAttributeMaxDynamicSharedMemorySize, SMEM_BYTES);
    rssm_scan<true><<<NB / 16, 256, SMEM_BYTES, stream>>>(embed, embBf, action, wsH, wsB, out);
  } else {
    (void)hipFuncSetAttribute((const void*)rssm_scan<false>,
                              hipFuncAttributeMaxDynamicSharedMemorySize, SMEM_BYTES);
    rssm_scan<false><<<NB / 16, 256, SMEM_BYTES, stream>>>(embed, embBf, action, wsH, wsB, out);
  }
}
